// GCN_34488587387573
// MI455X (gfx1250) — matrix operand order repaired and checked by reading, not compiled, not run on hardware
//
#include <hip/hip_runtime.h>
#include <hip/hip_bf16.h>

#define N_NODES 100000
#define N_EDGES 1600000
#define NFEAT   512
#define NHID    128
#define NCLASS  40
#define KST1    (NFEAT / 32)   // 16 K-steps in GEMM1
#define KST2    (NHID / 32)    // 4  K-steps in GEMM2
#define NT1     (NHID / 16)    // 8  N tiles in GEMM1
#define NT2     3              // 48-wide padded N in GEMM2

typedef __attribute__((ext_vector_type(16))) __bf16 v16bf;
typedef __attribute__((ext_vector_type(8)))  float  v8f;

__device__ __forceinline__ __bf16 f2bf(float f) {
    return static_cast<__bf16>(f);   // native cvt on gfx1250
}

// ---------------------------------------------------------------------------
// Pre-pack W1[512x128] into WMMA B-operand register order:
//   W1p[kst][nt][lane][i] = W1[(kst*32 + i + 16*(lane>>4)) * 128 + nt*16 + (lane&15)]
// Each GEMM lane then reads its 16 bf16 B elements as one contiguous 32B vector.
// ---------------------------------------------------------------------------
__global__ void gcn_pack_w1(const float* __restrict__ W1, __bf16* __restrict__ W1p) {
    const int idx = blockIdx.x * blockDim.x + threadIdx.x;   // 0 .. 65535
    if (idx >= KST1 * NT1 * 32 * 16) return;
    const int i    = idx & 15;
    const int lane = (idx >> 4) & 31;
    const int nt   = (idx >> 9) & 7;
    const int kst  = idx >> 12;
    const int K = kst * 32 + i + 16 * (lane >> 4);
    const int N = nt * 16 + (lane & 15);
    W1p[idx] = f2bf(W1[K * NHID + N]);
}

// Same for W2[128x40], padded to N=48 with zeros.
__global__ void gcn_pack_w2(const float* __restrict__ W2, __bf16* __restrict__ W2p) {
    const int idx = blockIdx.x * blockDim.x + threadIdx.x;   // 0 .. 6143
    if (idx >= KST2 * NT2 * 32 * 16) return;
    const int i    = idx & 15;
    const int lane = (idx >> 4) & 31;
    const int ntk  = idx >> 9;         // nt + 3*kst
    const int nt   = ntk % NT2;
    const int kst  = ntk / NT2;
    const int K = kst * 32 + i + 16 * (lane >> 4);
    const int N = nt * 16 + (lane & 15);
    W2p[idx] = (N < NCLASS) ? f2bf(W2[K * NCLASS + N]) : (__bf16)0.0f;
}

// Load + convert one 16x32 bf16 A tile (optionally fused ReLU).
// A-layout K runs are contiguous in one chunk [k0+16*hi, +16).
template <bool RELU>
__device__ __forceinline__ v16bf load_a_tile(const float* __restrict__ arow,
                                             int k0, int hi) {
    const float4* p0 = (const float4*)(arow + k0 + 16 * hi);
    const float4* p1 = (const float4*)(arow + k0 + 16 * hi + 8);
    float4 f0 = p0[0], f1 = p0[1], f2 = p1[0], f3 = p1[1];
    if (RELU) {
        f0.x=fmaxf(f0.x,0.f); f0.y=fmaxf(f0.y,0.f); f0.z=fmaxf(f0.z,0.f); f0.w=fmaxf(f0.w,0.f);
        f1.x=fmaxf(f1.x,0.f); f1.y=fmaxf(f1.y,0.f); f1.z=fmaxf(f1.z,0.f); f1.w=fmaxf(f1.w,0.f);
        f2.x=fmaxf(f2.x,0.f); f2.y=fmaxf(f2.y,0.f); f2.z=fmaxf(f2.z,0.f); f2.w=fmaxf(f2.w,0.f);
        f3.x=fmaxf(f3.x,0.f); f3.y=fmaxf(f3.y,0.f); f3.z=fmaxf(f3.z,0.f); f3.w=fmaxf(f3.w,0.f);
    }
    v16bf a;
    a[0]=f2bf(f0.x);  a[1]=f2bf(f0.y);  a[2]=f2bf(f0.z);  a[3]=f2bf(f0.w);
    a[4]=f2bf(f1.x);  a[5]=f2bf(f1.y);  a[6]=f2bf(f1.z);  a[7]=f2bf(f1.w);
    a[8]=f2bf(f2.x);  a[9]=f2bf(f2.y);  a[10]=f2bf(f2.z); a[11]=f2bf(f2.w);
    a[12]=f2bf(f3.x); a[13]=f2bf(f3.y); a[14]=f2bf(f3.z); a[15]=f2bf(f3.w);
    return a;
}

// ---------------------------------------------------------------------------
// GEMM1: S1[100000x128] = X[100000x512] @ W1 (pre-packed).
// One wave per 16-row M tile computes the full 128-wide N strip (8 WMMA tiles).
// ---------------------------------------------------------------------------
__global__ __launch_bounds__(32) void gcn_gemm1_wmma(
    const float* __restrict__ X, const __bf16* __restrict__ W1p,
    float* __restrict__ S1) {
    const int tile = blockIdx.x;
    const int lane = threadIdx.x;
    const int l15  = lane & 15;
    const int hi   = lane >> 4;
    const float* xrow = X + ((long)tile * 16 + l15) * NFEAT;

    v8f acc[NT1];
#pragma unroll
    for (int n = 0; n < NT1; ++n) acc[n] = (v8f){0.f,0.f,0.f,0.f,0.f,0.f,0.f,0.f};

    for (int kst = 0; kst < KST1; ++kst) {
        const v16bf a = load_a_tile<false>(xrow, kst * 32, hi);
#pragma unroll
        for (int nt = 0; nt < NT1; ++nt) {
            const v16bf b = *(const v16bf*)(W1p + (((kst * NT1 + nt) * 32) + lane) * 16);
            acc[nt] = __builtin_amdgcn_wmma_f32_16x16x32_bf16(
                false, a, false, b, (short)0, acc[nt], false, false);
        }
    }
#pragma unroll
    for (int nt = 0; nt < NT1; ++nt) {
        const int ncol = nt * 16 + l15;
#pragma unroll
        for (int v = 0; v < 8; ++v) {
            const long m = (long)tile * 16 + v + hi * 8;
            S1[m * NHID + ncol] = acc[nt][v];
        }
    }
}

// ---------------------------------------------------------------------------
// GEMM2: S2[100000x40] = relu(H) @ W2 (pre-packed, N padded to 48)
// ---------------------------------------------------------------------------
__global__ __launch_bounds__(32) void gcn_gemm2_wmma(
    const float* __restrict__ H, const __bf16* __restrict__ W2p,
    float* __restrict__ S2) {
    const int tile = blockIdx.x;
    const int lane = threadIdx.x;
    const int l15  = lane & 15;
    const int hi   = lane >> 4;
    const float* hrow = H + ((long)tile * 16 + l15) * NHID;

    v8f acc[NT2];
#pragma unroll
    for (int n = 0; n < NT2; ++n) acc[n] = (v8f){0.f,0.f,0.f,0.f,0.f,0.f,0.f,0.f};

#pragma unroll
    for (int kst = 0; kst < KST2; ++kst) {
        const v16bf a = load_a_tile<true>(hrow, kst * 32, hi);   // fused ReLU
#pragma unroll
        for (int nt = 0; nt < NT2; ++nt) {
            const v16bf b = *(const v16bf*)(W2p + (((kst * NT2 + nt) * 32) + lane) * 16);
            acc[nt] = __builtin_amdgcn_wmma_f32_16x16x32_bf16(
                false, a, false, b, (short)0, acc[nt], false, false);
        }
    }
#pragma unroll
    for (int nt = 0; nt < NT2; ++nt) {
        const int ncol = nt * 16 + l15;
        if (ncol < NCLASS) {
#pragma unroll
            for (int v = 0; v < 8; ++v) {
                const long m = (long)tile * 16 + v + hi * 8;
                S2[m * NCLASS + ncol] = acc[nt][v];
            }
        }
    }
}

// ---------------------------------------------------------------------------
// Init buffers with broadcast bias (re-run each call -> graph-replay safe)
// ---------------------------------------------------------------------------
__global__ void gcn_init_bias(float* __restrict__ dst, const float* __restrict__ bias,
                              int width, long total) {
    long gid = (long)blockIdx.x * blockDim.x + threadIdx.x;
    if (gid < total) dst[gid] = bias[gid % width];
}

// ---------------------------------------------------------------------------
// SpMM1: H[row] += val * S1[col]   (128 feats; 32 lanes/edge x float4)
// ---------------------------------------------------------------------------
__global__ void gcn_spmm1(const int* __restrict__ erow, const int* __restrict__ ecol,
                          const float* __restrict__ eval, const float* __restrict__ S1,
                          float* __restrict__ H) {
    const long gid = (long)blockIdx.x * blockDim.x + threadIdx.x;
    const long e = gid >> 5;
    if (e >= N_EDGES) return;
    const int part = (int)(gid & 31);
    const int r = erow[e], c = ecol[e];
    const float v = eval[e];
    const float4 g = *(const float4*)(S1 + (long)c * NHID + part * 4);
    float* dst = H + (long)r * NHID + part * 4;
    atomicAdd(dst + 0, v * g.x);
    atomicAdd(dst + 1, v * g.y);
    atomicAdd(dst + 2, v * g.z);
    atomicAdd(dst + 3, v * g.w);
}

// ---------------------------------------------------------------------------
// SpMM2: OUT[row] += val * S2[col]  (40 feats; 10 lanes/edge x float4)
// ---------------------------------------------------------------------------
__global__ void gcn_spmm2(const int* __restrict__ erow, const int* __restrict__ ecol,
                          const float* __restrict__ eval, const float* __restrict__ S2,
                          float* __restrict__ out) {
    const long gid = (long)blockIdx.x * blockDim.x + threadIdx.x;
    const long e = gid / 10;
    if (e >= N_EDGES) return;
    const int part = (int)(gid - e * 10);
    const int r = erow[e], c = ecol[e];
    const float v = eval[e];
    const float4 g = *(const float4*)(S2 + (long)c * NCLASS + part * 4);
    float* dst = out + (long)r * NCLASS + part * 4;
    atomicAdd(dst + 0, v * g.x);
    atomicAdd(dst + 1, v * g.y);
    atomicAdd(dst + 2, v * g.z);
    atomicAdd(dst + 3, v * g.w);
}

// ---------------------------------------------------------------------------
// Row-wise log_softmax over 40 classes, in-place
// ---------------------------------------------------------------------------
__global__ void gcn_log_softmax(float* __restrict__ out) {
    const long node = (long)blockIdx.x * blockDim.x + threadIdx.x;
    if (node >= N_NODES) return;
    float* p = out + node * NCLASS;
    float mx = -3.402823466e+38f;
#pragma unroll
    for (int i = 0; i < NCLASS; ++i) mx = fmaxf(mx, p[i]);
    float s = 0.f;
#pragma unroll
    for (int i = 0; i < NCLASS; ++i) s += __expf(p[i] - mx);
    const float lse = mx + __logf(s);
#pragma unroll
    for (int i = 0; i < NCLASS; ++i) p[i] = p[i] - lse;
}

extern "C" void kernel_launch(void* const* d_in, const int* in_sizes, int n_in,
                              void* d_out, int out_size, void* d_ws, size_t ws_size,
                              hipStream_t stream) {
    const float* X    = (const float*)d_in[0];
    const int*   erow = (const int*)  d_in[1];
    const int*   ecol = (const int*)  d_in[2];
    const float* eval = (const float*)d_in[3];
    const float* W1   = (const float*)d_in[4];
    const float* b1   = (const float*)d_in[5];
    const float* W2   = (const float*)d_in[6];
    const float* b2   = (const float*)d_in[7];
    float* out = (float*)d_out;

    float* S1 = (float*)d_ws;                          // 100000*128 f32 = 51.2 MB
    float* H  = S1 + (size_t)N_NODES * NHID;           // 51.2 MB
    float* S2 = H  + (size_t)N_NODES * NHID;           // 16 MB
    __bf16* W1p = (__bf16*)(S2 + (size_t)N_NODES * NCLASS);  // 128 KB
    __bf16* W2p = W1p + (size_t)KST1 * NT1 * 32 * 16;        // 12 KB

    const int mtiles = N_NODES / 16;                   // 6250

    // 0) pre-pack weights into WMMA B-register order
    gcn_pack_w1<<<(KST1 * NT1 * 32 * 16 + 255) / 256, 256, 0, stream>>>(W1, W1p);
    gcn_pack_w2<<<(KST2 * NT2 * 32 * 16 + 255) / 256, 256, 0, stream>>>(W2, W2p);

    // 1) S1 = X @ W1 (WMMA bf16 -> f32 accum)
    gcn_gemm1_wmma<<<mtiles, 32, 0, stream>>>(X, W1p, S1);

    // 2) H = broadcast(b1)
    {
        const long total = (long)N_NODES * NHID;
        gcn_init_bias<<<(int)((total + 255) / 256), 256, 0, stream>>>(H, b1, NHID, total);
    }

    // 3) H += A @ S1 (edge-parallel atomics)
    {
        const long work = (long)N_EDGES * 32;
        gcn_spmm1<<<(int)((work + 255) / 256), 256, 0, stream>>>(erow, ecol, eval, S1, H);
    }

    // 4) S2 = relu(H) @ W2 (WMMA, fused ReLU)
    gcn_gemm2_wmma<<<mtiles, 32, 0, stream>>>(H, W2p, S2);

    // 5) out = broadcast(b2)
    {
        const long total = (long)N_NODES * NCLASS;
        gcn_init_bias<<<(int)((total + 255) / 256), 256, 0, stream>>>(out, b2, NCLASS, total);
    }

    // 6) out += A @ S2
    {
        const long work = (long)N_EDGES * 10;
        gcn_spmm2<<<(int)((work + 255) / 256), 256, 0, stream>>>(erow, ecol, eval, S2, out);
    }

    // 7) log_softmax in-place
    gcn_log_softmax<<<(N_NODES + 255) / 256, 256, 0, stream>>>(out);
}
